// myRNN_51616916963643
// MI455X (gfx1250) — matrix operand order repaired and checked by reading, not compiled, not run on hardware
//
#include <hip/hip_runtime.h>

#define B_   32
#define T_   512
#define D_   512
#define H_   512
#define G4_  2048   // 4*H

typedef __attribute__((ext_vector_type(16))) __bf16 v16bf;
typedef __attribute__((ext_vector_type(8)))  float  v8f;

union FragAB { v16bf v; uint4 u[2]; };

__device__ __forceinline__ unsigned short f2bf(float f) {
    unsigned u = __float_as_uint(f);
    unsigned r = u + 0x7FFFu + ((u >> 16) & 1u);   // round-to-nearest-even
    return (unsigned short)(r >> 16);
}
__device__ __forceinline__ float bf2f(unsigned short h) {
    return __uint_as_float(((unsigned)h) << 16);
}
__device__ __forceinline__ float sigmoidf_(float x) {
    return 1.0f / (1.0f + __expf(-x));
}
__device__ __forceinline__ float tanhf_(float x) {
    float e = __expf(2.0f * x);
    return 1.0f - 2.0f / (e + 1.0f);   // (e-1)/(e+1)
}

// ---------------- init: zero barrier + h ping-pong ----------------
__global__ void k_init(unsigned* __restrict__ bar, unsigned short* __restrict__ hbuf) {
    int i = blockIdx.x * blockDim.x + threadIdx.x;
    if (i < 2) bar[i] = 0u;
    const int n = 2 * B_ * H_;
    for (int j = i; j < n; j += gridDim.x * blockDim.x) hbuf[j] = 0;
}

// ---------------- pack x: [B][T][D] f32 -> [T*B][D] bf16 ----------------
__global__ void k_pack_x(const float* __restrict__ x, unsigned short* __restrict__ xbf) {
    long i = (long)blockIdx.x * blockDim.x + threadIdx.x;
    if (i >= (long)B_ * T_ * D_) return;
    int  k = (int)(i % D_);
    long r = i / D_;
    int  t = (int)(r % T_);
    int  b = (int)(r / T_);
    xbf[((long)t * B_ + b) * D_ + k] = f2bf(x[i]);
}

// ------- pack W0/W1 [1024][2048] f32 -> four [2048][512] bf16 (N-major) -------
__global__ void k_pack_w(const float* __restrict__ W0, const float* __restrict__ W1,
                         unsigned short* __restrict__ w0x, unsigned short* __restrict__ w0h,
                         unsigned short* __restrict__ w1x, unsigned short* __restrict__ w1h) {
    long i = (long)blockIdx.x * blockDim.x + threadIdx.x;
    if (i >= 4L * G4_ * 512) return;
    int  k = (int)(i % 512);
    long r = i / 512;
    int  n = (int)(r % G4_);
    int  which = (int)(r / G4_);
    size_t dst = (size_t)n * 512 + k;
    if      (which == 0) w0x[dst] = f2bf(W0[(size_t)k         * G4_ + n]);
    else if (which == 1) w0h[dst] = f2bf(W0[(size_t)(512 + k) * G4_ + n]);
    else if (which == 2) w1x[dst] = f2bf(W1[(size_t)k         * G4_ + n]);
    else                 w1h[dst] = f2bf(W1[(size_t)(512 + k) * G4_ + n]);
}

// ------- batched gate GEMM: C[M][2048] = Abf[M][512] @ Wt[2048][512]^T + bias -------
// grid (M/32, 2048/64), one wave per block; wave computes a 32x64 f32 tile.
__global__ __launch_bounds__(32) void k_gemm_gates(
    const unsigned short* __restrict__ Abf,
    const unsigned short* __restrict__ Wt,
    const float* __restrict__ bias,
    float* __restrict__ C) {
    const int lane = threadIdx.x & 31;
    const int half = lane >> 4, ll = lane & 15;
    const int m0 = blockIdx.x * 32;
    const int n0 = blockIdx.y * 64;

    v8f acc[2][4];
    #pragma unroll
    for (int ni = 0; ni < 4; ++ni) {
        float bv = bias[n0 + ni * 16 + ll];
        #pragma unroll
        for (int mi = 0; mi < 2; ++mi)
            #pragma unroll
            for (int r = 0; r < 8; ++r) acc[mi][ni][r] = bv;
    }

    for (int kb = 0; kb < 512; kb += 32) {
        FragAB a[2], b;
        #pragma unroll
        for (int mi = 0; mi < 2; ++mi) {
            const unsigned short* ap = Abf + (size_t)(m0 + mi * 16 + ll) * 512;
            a[mi].u[0] = *(const uint4*)(ap + kb + half * 16);
            a[mi].u[1] = *(const uint4*)(ap + kb + half * 16 + 8);
        }
        #pragma unroll
        for (int ni = 0; ni < 4; ++ni) {
            const unsigned short* bp = Wt + (size_t)(n0 + ni * 16 + ll) * 512 + kb + half * 16;
            b.u[0] = *(const uint4*)(bp);
            b.u[1] = *(const uint4*)(bp + 8);
            #pragma unroll
            for (int mi = 0; mi < 2; ++mi)
                acc[mi][ni] = __builtin_amdgcn_wmma_f32_16x16x32_bf16(
                    false, a[mi].v, false, b.v, (short)0, acc[mi][ni], false, false);
        }
    }

    #pragma unroll
    for (int mi = 0; mi < 2; ++mi)
        #pragma unroll
        for (int ni = 0; ni < 4; ++ni)
            #pragma unroll
            for (int r = 0; r < 8; ++r) {
                int m = m0 + mi * 16 + r + 8 * half;
                int n = n0 + ni * 16 + ll;
                C[(size_t)m * G4_ + n] = acc[mi][ni][r];
            }
}

// ---------------- persistent recurrent layer ----------------
// grid = 32 blocks x 256 threads (8 waves = 2 M-tiles x 4 K-chunks).
// Block nb owns h-columns [nb*16, nb*16+16) across all 4 gates.
// Wave (mt,kc) computes the K-chunk [kc*128, kc*128+128) of all 4 gate tiles for
// M-rows [mt*16, mt*16+16); partials reduced via LDS; kc==0 does the gate math
// and keeps the cell state c in registers across all T steps.
__global__ __launch_bounds__(256) void k_lstm_layer(
    const float* __restrict__ Ag,          // [T*B][2048] input-proj + bias
    const unsigned short* __restrict__ Wh, // [2048][512] bf16, N-major
    unsigned short* __restrict__ Hout,     // [T*B][512] bf16
    unsigned short* __restrict__ hbuf,     // [2][B][512] bf16 ping-pong
    unsigned* __restrict__ bar) {          // {count, gen}
    __shared__ unsigned short lWh[4 * 16 * 512];       // 64 KB weight slice
    __shared__ float lred[6 * 4 * 32 * 8];             // 24 KB partial sums [p][g][lane][r]

    const int nb = blockIdx.x;
    const int n0 = nb * 16;
    const int tid = threadIdx.x;
    const int wave = tid >> 5;
    const int kc = wave & 3;        // K-chunk 0..3
    const int mt = wave >> 2;       // M-tile 0..1
    const int lane = tid & 31;
    const int half = lane >> 4, ll = lane & 15;

    // Preload this block's Wh slice (4 gates x 16 cols x 512 K) into LDS, once.
    {
        int row = tid >> 2, part = tid & 3;            // 64 rows x 4 quarters
        const uint4* src = (const uint4*)(Wh + (size_t)((row >> 4) * 512 + n0 + (row & 15)) * 512) + part * 16;
        uint4* dst = (uint4*)(lWh + (size_t)row * 512) + part * 16;
        #pragma unroll 4
        for (int j = 0; j < 16; ++j) dst[j] = src[j];
    }
    __syncthreads();

    v8f c;
    #pragma unroll
    for (int r = 0; r < 8; ++r) c[r] = 0.0f;

    for (int t = 0; t < T_; ++t) {
        const unsigned short* hin = hbuf + (size_t)(t & 1) * B_ * H_;
        unsigned short* hnext = hbuf + (size_t)((t + 1) & 1) * B_ * H_;

        // Issue the input-projection loads early (independent of h): full overlap
        // with the WMMA chain below.
        float agv[4][8];
        if (kc == 0) {
            const float* abase = Ag + (size_t)t * B_ * G4_;
            #pragma unroll
            for (int r = 0; r < 8; ++r) {
                const float* az = abase + (size_t)(mt * 16 + r + 8 * half) * G4_ + n0 + ll;
                agv[0][r] = az[0 * H_];
                agv[1][r] = az[1 * H_];
                agv[2][r] = az[2 * H_];
                agv[3][r] = az[3 * H_];
            }
        }

        v8f acc[4];
        #pragma unroll
        for (int g = 0; g < 4; ++g)
            #pragma unroll
            for (int r = 0; r < 8; ++r) acc[g][r] = 0.0f;

        const int kbeg = kc * 128;
        #pragma unroll
        for (int ki = 0; ki < 4; ++ki) {
            const int kb = kbeg + ki * 32;
            FragAB a, b;
            const unsigned short* ap = hin + (size_t)(mt * 16 + ll) * 512;
            a.u[0] = *(const uint4*)(ap + kb + half * 16);
            a.u[1] = *(const uint4*)(ap + kb + half * 16 + 8);
            #pragma unroll
            for (int g = 0; g < 4; ++g) {
                const unsigned short* bp = lWh + (size_t)(g * 16 + ll) * 512 + kb + half * 16;
                b.u[0] = *(const uint4*)(bp);
                b.u[1] = *(const uint4*)(bp + 8);
                acc[g] = __builtin_amdgcn_wmma_f32_16x16x32_bf16(
                    false, a.v, false, b.v, (short)0, acc[g], false, false);
            }
        }

        // Cross-wave K reduction through LDS.
        if (kc > 0) {
            int p = mt * 3 + (kc - 1);
            #pragma unroll
            for (int g = 0; g < 4; ++g) {
                float4* dst = (float4*)&lred[(((p * 4 + g) * 32) + lane) * 8];
                union { v8f v; float4 f[2]; } tmp; tmp.v = acc[g];
                dst[0] = tmp.f[0];
                dst[1] = tmp.f[1];
            }
        }
        __syncthreads();

        if (kc == 0) {
            #pragma unroll
            for (int p3 = 0; p3 < 3; ++p3) {
                int p = mt * 3 + p3;
                #pragma unroll
                for (int g = 0; g < 4; ++g) {
                    const float4* src = (const float4*)&lred[(((p * 4 + g) * 32) + lane) * 8];
                    float4 s0 = src[0], s1 = src[1];
                    acc[g][0] += s0.x; acc[g][1] += s0.y; acc[g][2] += s0.z; acc[g][3] += s0.w;
                    acc[g][4] += s1.x; acc[g][5] += s1.y; acc[g][6] += s1.z; acc[g][7] += s1.w;
                }
            }

            // Gate math + state update (registers only), then publish new h.
            #pragma unroll
            for (int r = 0; r < 8; ++r) {
                int m = mt * 16 + r + 8 * half;
                int col = n0 + ll;
                float zf = acc[0][r] + agv[0][r];
                float zi = acc[1][r] + agv[1][r];
                float zg = acc[2][r] + agv[2][r];
                float zo = acc[3][r] + agv[3][r];
                float nh = sigmoidf_(zo) * tanhf_(c[r]);          // tanh of OLD cell state
                c[r] = c[r] * sigmoidf_(zf) + tanhf_(zg) * sigmoidf_(zi);
                unsigned short hb = f2bf(nh);
                hnext[(size_t)m * H_ + col] = hb;
                Hout[((size_t)t * B_ + m) * H_ + col] = hb;
            }
        }

        // device-wide barrier (32 resident blocks); also orders lred reuse.
        __threadfence();
        __syncthreads();
        if (tid == 0) {
            unsigned g = atomicAdd(bar + 1, 0u);       // sample gen before arriving
            unsigned prev = atomicAdd(bar + 0, 1u);
            if (prev == (unsigned)(gridDim.x - 1)) {
                atomicExch(bar + 0, 0u);
                atomicAdd(bar + 1, 1u);                // release
            } else {
                while (atomicAdd(bar + 1, 0u) == g) { /* spin */ }
            }
        }
        __syncthreads();
        __threadfence();
    }
}

// ---------------- projection: out[b][t][2] = H2[t][b][:] @ Wr + br ----------------
__global__ void k_proj(const unsigned short* __restrict__ H2,
                       const float* __restrict__ Wr, const float* __restrict__ br,
                       float* __restrict__ out) {
    int i = blockIdx.x * blockDim.x + threadIdx.x;
    if (i >= B_ * T_) return;
    int t = i % T_, b = i / T_;
    const unsigned short* h = H2 + ((size_t)t * B_ + b) * H_;
    float a0 = br[0], a1 = br[1];
    for (int k = 0; k < H_; k += 8) {
        uint4 hv4 = *(const uint4*)(h + k);
        const unsigned* hw = (const unsigned*)&hv4;
        #pragma unroll
        for (int j = 0; j < 4; ++j) {
            float h0 = __uint_as_float(hw[j] << 16);
            float h1 = __uint_as_float(hw[j] & 0xFFFF0000u);
            a0 += h0 * Wr[(k + 2 * j) * 2 + 0];
            a1 += h0 * Wr[(k + 2 * j) * 2 + 1];
            a0 += h1 * Wr[(k + 2 * j + 1) * 2 + 0];
            a1 += h1 * Wr[(k + 2 * j + 1) * 2 + 1];
        }
    }
    out[(size_t)i * 2 + 0] = a0;
    out[(size_t)i * 2 + 1] = a1;
}

extern "C" void kernel_launch(void* const* d_in, const int* in_sizes, int n_in,
                              void* d_out, int out_size, void* d_ws, size_t ws_size,
                              hipStream_t stream) {
    (void)in_sizes; (void)n_in; (void)out_size; (void)ws_size;
    const float* x  = (const float*)d_in[0];
    const float* W0 = (const float*)d_in[1];
    const float* b0 = (const float*)d_in[2];
    const float* W1 = (const float*)d_in[3];
    const float* b1 = (const float*)d_in[4];
    const float* Wr = (const float*)d_in[5];
    const float* br = (const float*)d_in[6];
    float* out = (float*)d_out;

    char* ws = (char*)d_ws;
    size_t off = 0;
    auto carve = [&](size_t bytes) { char* p = ws + off; off += (bytes + 255) & ~(size_t)255; return p; };

    unsigned*       bar  = (unsigned*)      carve(256);
    unsigned short* hbuf = (unsigned short*)carve((size_t)2 * B_ * H_ * 2);
    unsigned short* xbf  = (unsigned short*)carve((size_t)B_ * T_ * D_ * 2);
    unsigned short* w0x  = (unsigned short*)carve((size_t)G4_ * 512 * 2);
    unsigned short* w0h  = (unsigned short*)carve((size_t)G4_ * 512 * 2);
    unsigned short* w1x  = (unsigned short*)carve((size_t)G4_ * 512 * 2);
    unsigned short* w1h  = (unsigned short*)carve((size_t)G4_ * 512 * 2);
    float*          Ag   = (float*)         carve((size_t)T_ * B_ * G4_ * 4);  // reused by both layers
    unsigned short* H1   = (unsigned short*)carve((size_t)T_ * B_ * H_ * 2);
    unsigned short* H2   = (unsigned short*)carve((size_t)T_ * B_ * H_ * 2);

    const long nx = (long)B_ * T_ * D_;
    const long nw = 4L * G4_ * 512;

    // 1) init barrier + h ping-pong
    k_init<<<1, 256, 0, stream>>>(bar, hbuf);
    // 2) pack activations and weights to bf16 / N-major
    k_pack_x<<<(int)((nx + 255) / 256), 256, 0, stream>>>(x, xbf);
    k_pack_w<<<(int)((nw + 255) / 256), 256, 0, stream>>>(W0, W1, w0x, w0h, w1x, w1h);
    // 3) A0 = Xbf @ W0x + b0   (batched input projection, layer 1)
    k_gemm_gates<<<dim3((T_ * B_) / 32, G4_ / 64), 32, 0, stream>>>(xbf, w0x, b0, Ag);
    // 4) layer-1 recurrence (persistent, device barrier per step)
    k_lstm_layer<<<32, 256, 0, stream>>>(Ag, w0h, H1, hbuf, bar);
    // 5) reset state for layer 2
    k_init<<<1, 256, 0, stream>>>(bar, hbuf);
    // 6) A1 = H1 @ W1x + b1   (batched input projection, layer 2)
    k_gemm_gates<<<dim3((T_ * B_) / 32, G4_ / 64), 32, 0, stream>>>(H1, w1x, b1, Ag);
    // 7) layer-2 recurrence
    k_lstm_layer<<<32, 256, 0, stream>>>(Ag, w1h, H2, hbuf, bar);
    // 8) output projection
    k_proj<<<(B_ * T_ + 255) / 256, 256, 0, stream>>>(H2, Wr, br, out);
}
